// RPPAttenLayer_69088843923830
// MI455X (gfx1250) — matrix operand order repaired and checked by reading, not compiled, not run on hardware
//
#include <hip/hip_runtime.h>
#include <hip/hip_bf16.h>

typedef __attribute__((ext_vector_type(16))) __bf16 v16bf;
typedef __attribute__((ext_vector_type(8)))  __bf16 v8bf;
typedef __attribute__((ext_vector_type(8)))  float  v8f;

#define BB 2
#define N1S 8192
#define N2S 8192
#define CC 128
#define NSAMP 16
#define WSTR 136   // padded row stride (halfs): 272 B -> bank stagger of 4, 16-B aligned

__device__ __forceinline__ float leakyf(float x) { return x >= 0.f ? x : 0.1f * x; }

// Build a 16-half WMMA fragment from two 16-byte LDS runs.
__device__ __forceinline__ v16bf ld_frag16(const __bf16* p0, const __bf16* p1) {
  v8bf lo = *(const v8bf*)p0;
  v8bf hi = *(const v8bf*)p1;
  v16bf r;
  #pragma unroll
  for (int h = 0; h < 8; ++h) { r[h] = lo[h]; r[h + 8] = hi[h]; }
  return r;
}

// ---------------------------------------------------------------------------
// Kernel 1: positional encoder (conv1d 3->128) + GroupNorm(8 groups) + leaky,
// added to features. One block per (batch, group).
// ---------------------------------------------------------------------------
__global__ __launch_bounds__(256) void posenc_kernel(
    const float* __restrict__ xyz, const float* __restrict__ feat_in,
    float* __restrict__ feat_out,
    const float* __restrict__ pw, const float* __restrict__ pb,
    const float* __restrict__ pg, const float* __restrict__ pbeta, int N)
{
  int b = blockIdx.x >> 3;
  int g = blockIdx.x & 7;
  int tid = threadIdx.x;
  const float* X = xyz + (size_t)b * 3 * N;
  int total = 16 * N;

  float sum = 0.f, sq = 0.f;
  for (int e = tid; e < total; e += 256) {
    int cl = e / N;
    int n  = e - cl * N;
    int c  = g * 16 + cl;
    float h = pw[c*3+0]*X[n] + pw[c*3+1]*X[N+n] + pw[c*3+2]*X[2*N+n] + pb[c];
    sum += h; sq += h * h;
  }
  #pragma unroll
  for (int o = 16; o >= 1; o >>= 1) {
    sum += __shfl_xor(sum, o, 32);
    sq  += __shfl_xor(sq,  o, 32);
  }
  __shared__ float rs[8], rq[8];
  __shared__ float smu, srstd;
  if ((tid & 31) == 0) { rs[tid >> 5] = sum; rq[tid >> 5] = sq; }
  __syncthreads();
  if (tid == 0) {
    float S = 0.f, Q = 0.f;
    for (int i = 0; i < 8; ++i) { S += rs[i]; Q += rq[i]; }
    float mu = S / (float)total;
    float var = Q / (float)total - mu * mu;
    smu = mu; srstd = rsqrtf(var + 1e-5f);
  }
  __syncthreads();
  float mu = smu, rstd = srstd;
  for (int e = tid; e < total; e += 256) {
    int cl = e / N;
    int n  = e - cl * N;
    int c  = g * 16 + cl;
    float h = pw[c*3+0]*X[n] + pw[c*3+1]*X[N+n] + pw[c*3+2]*X[2*N+n] + pb[c];
    float v = leakyf((h - mu) * rstd * pg[c] + pbeta[c]);
    size_t o_ = (size_t)b * CC * N + (size_t)c * N + n;
    feat_out[o_] = feat_in[o_] + v;
  }
}

// ---------------------------------------------------------------------------
// Kernel 2: kNN (top-16 smallest squared distances). xyz2 of one batch staged
// fully in LDS (96 KB). One thread per query, unrolled insertion top-16.
// ---------------------------------------------------------------------------
__global__ __launch_bounds__(256) void knn_kernel(
    const float* __restrict__ xyz1, const float* __restrict__ xyz2,
    int* __restrict__ idx, int n1, int n2)
{
  extern __shared__ char smem[];
  float* sxyz = (float*)smem; // [n2][3]
  int nblk = n1 / 256;
  int b  = blockIdx.x / nblk;
  int q0 = (blockIdx.x % nblk) * 256;
  const float* X2 = xyz2 + (size_t)b * 3 * n2;
  for (int i = threadIdx.x; i < n2; i += 256) {
    sxyz[i*3+0] = X2[i];
    sxyz[i*3+1] = X2[n2 + i];
    sxyz[i*3+2] = X2[2*n2 + i];
  }
  __syncthreads();

  int n = q0 + threadIdx.x;
  const float* X1 = xyz1 + (size_t)b * 3 * n1;
  float qx = X1[n], qy = X1[n1 + n], qz = X1[2*n1 + n];

  float bd[NSAMP]; int bi[NSAMP];
  #pragma unroll
  for (int t = 0; t < NSAMP; ++t) { bd[t] = 3.4e38f; bi[t] = 0; }

  for (int j = 0; j < n2; ++j) {
    float dx = sxyz[j*3+0] - qx;
    float dy = sxyz[j*3+1] - qy;
    float dz = sxyz[j*3+2] - qz;
    float d = dx*dx + dy*dy + dz*dz;
    if (d < bd[NSAMP-1]) {
      int pos = NSAMP - 1;
      #pragma unroll
      for (int t = NSAMP - 2; t >= 0; --t) {
        if (bd[t] > d) { bd[t+1] = bd[t]; bi[t+1] = bi[t]; pos = t; }
      }
      bd[pos] = d; bi[pos] = j;
    }
  }
  size_t o_ = ((size_t)b * n1 + n) * NSAMP;
  #pragma unroll
  for (int t = 0; t < NSAMP; ++t) idx[o_ + t] = bi[t];
}

// ---------------------------------------------------------------------------
// Kernel 3: fused gather + encoders + q/v WMMA GEMMs + softmax attention.
// 4 waves/block, one query per wave. Weights row-stride padded to WSTR halfs;
// per-query X matrices stored transposed [s][k] so every WMMA fragment is two
// 16-byte LDS loads (ds_load_b128) with bank-staggered rows.
// ---------------------------------------------------------------------------
__global__ __launch_bounds__(128) void atten_kernel(
    const float* __restrict__ xyz1, const float* __restrict__ xyz2,
    const float* __restrict__ feat1p, const float* __restrict__ feat2p,
    const int* __restrict__ idx,
    const float* __restrict__ ek_w, const float* __restrict__ ek_b,
    const float* __restrict__ eq_w, const float* __restrict__ eq_b,
    const float* __restrict__ ev1_w, const float* __restrict__ ev1_b,
    const float* __restrict__ ev2_w, const float* __restrict__ ev2_b,
    const float* __restrict__ qk_w, const float* __restrict__ v_w,
    float* __restrict__ featnew,
    int n1, int n2, int nQB)
{
  extern __shared__ char smem[];
  __bf16* sQK  = (__bf16*)smem;              // [128][WSTR] (34816 B)
  __bf16* sVW  = sQK + CC*WSTR;              // [128][WSTR] (34816 B)
  float*  sEnc = (float*)(sVW + CC*WSTR);    // 4 encoders * [c][w0,w1,w2,b] (8 KB)
  char* pwv = (char*)(sEnc + 4*512);
  const int PERWAVE = 4 * (16*WSTR*2) + 256; // XQ,XV,KM,E2 [16][WSTR] bf16 + aux
  int wv = threadIdx.x >> 5, lane = threadIdx.x & 31;
  __bf16* sXQ = (__bf16*)(pwv + wv * PERWAVE);
  __bf16* sXV = sXQ + 16*WSTR;
  __bf16* sKM = sXV + 16*WSTR;
  __bf16* sE2 = sKM + 16*WSTR;
  float*  sAux = (float*)(sE2 + 16*WSTR);    // rel[3][16] floats + idx[16] ints
  int*    sJd  = (int*)(sAux + 48);

  // Stage weights once per block (row-padded).
  for (int i = threadIdx.x; i < CC*CC; i += 128) {
    int r = i >> 7, k = i & 127;
    sQK[r*WSTR + k] = (__bf16)qk_w[i];
    sVW[r*WSTR + k] = (__bf16)v_w[i];
  }
  {
    int c = threadIdx.x; // blockDim == 128 == CC
    sEnc[0*512 + c*4+0] = ek_w[c*3+0];  sEnc[0*512 + c*4+1] = ek_w[c*3+1];
    sEnc[0*512 + c*4+2] = ek_w[c*3+2];  sEnc[0*512 + c*4+3] = ek_b[c];
    sEnc[1*512 + c*4+0] = eq_w[c*3+0];  sEnc[1*512 + c*4+1] = eq_w[c*3+1];
    sEnc[1*512 + c*4+2] = eq_w[c*3+2];  sEnc[1*512 + c*4+3] = eq_b[c];
    sEnc[2*512 + c*4+0] = ev1_w[c*3+0]; sEnc[2*512 + c*4+1] = ev1_w[c*3+1];
    sEnc[2*512 + c*4+2] = ev1_w[c*3+2]; sEnc[2*512 + c*4+3] = ev1_b[c];
    sEnc[3*512 + c*4+0] = ev2_w[c*3+0]; sEnc[3*512 + c*4+1] = ev2_w[c*3+1];
    sEnc[3*512 + c*4+2] = ev2_w[c*3+2]; sEnc[3*512 + c*4+3] = ev2_b[c];
  }
  __syncthreads();

  int half = (lane < 16) ? 0 : 1;
  int sj = lane & 15;

  for (int qb = blockIdx.x; qb < nQB; qb += gridDim.x) {
    int qglob = qb * 4 + wv;
    int b = qglob / n1, n = qglob - b * n1;

    // Prefetch next iteration's neighbor indices (global_prefetch_b8).
    if (qb + (int)gridDim.x < nQB)
      __builtin_prefetch(&idx[((size_t)qb + gridDim.x) * 4 * NSAMP], 0, 1);

    // Phase A1: relative xyz + neighbor indices (lanes 0..15).
    if (lane < 16) {
      int j = idx[(size_t)qglob * NSAMP + lane];
      sJd[lane] = j;
      const float* X1 = xyz1 + (size_t)b * 3 * n1;
      const float* X2 = xyz2 + (size_t)b * 3 * n2;
      sAux[ 0 + lane] = X2[j]        - X1[n];
      sAux[16 + lane] = X2[n2 + j]   - X1[n1 + n];
      sAux[32 + lane] = X2[2*n2 + j] - X1[2*n1 + n];
    }
    __syncthreads();

    // Phase A2: build XQ=feat1*eqc, XV=feat2_s*ev1c, KM=feat2_s*ekc, E2=ev2c,
    // stored transposed [s][k], c fastest across lanes.
    for (int t = lane; t < 2048; t += 32) {
      int c = t & 127, s = t >> 7;
      int j = sJd[s];
      float r0 = sAux[s], r1 = sAux[16+s], r2 = sAux[32+s];
      const float* E = sEnc + c*4;
      float ek = leakyf(E[0]   *r0 + E[1]   *r1 + E[2]   *r2 + E[3]);
      float eq = leakyf(E[512] *r0 + E[513] *r1 + E[514] *r2 + E[515]);
      float e1 = leakyf(E[1024]*r0 + E[1025]*r1 + E[1026]*r2 + E[1027]);
      float e2 = leakyf(E[1536]*r0 + E[1537]*r1 + E[1538]*r2 + E[1539]);
      float f2 = feat2p[(size_t)b*CC*n2 + (size_t)c*n2 + j];
      float f1 = feat1p[(size_t)b*CC*n1 + (size_t)c*n1 + n];
      int o = s*WSTR + c;
      sXQ[o] = (__bf16)(f1 * eq);
      sXV[o] = (__bf16)(f2 * e1);
      sKM[o] = (__bf16)(f2 * ek);
      sE2[o] = (__bf16)(e2);
    }
    __syncthreads();

    // Phase B: q = qk_w @ XQ via WMMA; accumulate q*k partial per lane.
    float apart = 0.f;
    for (int mt = 0; mt < 8; ++mt) {
      int moff = mt * 16;
      v8f cq = {};
      for (int kt = 0; kt < 4; ++kt) {
        int koff = kt * 32;
        const __bf16* pA = sQK + (moff + sj)*WSTR + koff + half*8;
        v16bf a = ld_frag16(pA, pA + 16);
        const __bf16* pB = sXQ + sj*WSTR + koff + half*8;
        v16bf bq = ld_frag16(pB, pB + 16);
        cq = __builtin_amdgcn_wmma_f32_16x16x32_bf16(false, a, false, bq,
                                                     (short)0, cq, false, false);
      }
      v8bf km = *(const v8bf*)(sKM + sj*WSTR + moff + half*8);
      #pragma unroll
      for (int r = 0; r < 8; ++r) apart += cq[r] * (float)km[r];
    }

    // Phase C: softmax over the 16 samples (wave32 shuffles).
    apart += __shfl_xor(apart, 16, 32);
    float av = apart * 0.088388347648318447f; // 1/sqrt(128)
    float mx = av;
    #pragma unroll
    for (int o = 8; o >= 1; o >>= 1) mx = fmaxf(mx, __shfl_xor(mx, o, 32));
    float ex = __expf(av - mx);
    float ssum = ex;
    #pragma unroll
    for (int o = 8; o >= 1; o >>= 1) ssum += __shfl_xor(ssum, o, 32);
    float at = ex / ssum;

    // Phase D: v = v_w @ XV via WMMA; feat_new = sum_s v*E2*atten + feat1.
    for (int mt = 0; mt < 8; ++mt) {
      int moff = mt * 16;
      v8f cv = {};
      for (int kt = 0; kt < 4; ++kt) {
        int koff = kt * 32;
        const __bf16* pA = sVW + (moff + sj)*WSTR + koff + half*8;
        v16bf a = ld_frag16(pA, pA + 16);
        const __bf16* pB = sXV + sj*WSTR + koff + half*8;
        v16bf bv = ld_frag16(pB, pB + 16);
        cv = __builtin_amdgcn_wmma_f32_16x16x32_bf16(false, a, false, bv,
                                                     (short)0, cv, false, false);
      }
      v8bf e2v = *(const v8bf*)(sE2 + sj*WSTR + moff + half*8);
      #pragma unroll
      for (int r = 0; r < 8; ++r) {
        int M = moff + half * 8 + r;
        float v = cv[r] * (float)e2v[r] * at;
        #pragma unroll
        for (int o = 8; o >= 1; o >>= 1) v += __shfl_xor(v, o, 32);
        if (sj == 0) {
          size_t o_ = (size_t)b*CC*n1 + (size_t)M*n1 + n;
          featnew[o_] = v + feat1p[o_];
        }
      }
    }
    __syncthreads();
  }
}

// ---------------------------------------------------------------------------
// Kernel 4: h = ff_w @ feat_new + ff_b via WMMA. 64 columns per block, 4 waves.
// Both operands in padded, fragment-contiguous LDS layouts.
// ---------------------------------------------------------------------------
__global__ __launch_bounds__(128) void ff_gemm_kernel(
    const float* __restrict__ featnew, const float* __restrict__ ff_w,
    const float* __restrict__ ff_b, float* __restrict__ hbuf, int n1)
{
  extern __shared__ char smem[];
  __bf16* sW = (__bf16*)smem;     // [128][WSTR]
  __bf16* sX = sW + CC*WSTR;      // [64][WSTR] transposed (col-major K-contig)
  int tid = threadIdx.x, wv = tid >> 5, lane = tid & 31;
  int nblk = n1 / 64;
  int b  = blockIdx.x / nblk;
  int n0 = (blockIdx.x % nblk) * 64;

  for (int i = tid; i < CC*CC; i += 128) {
    int r = i >> 7, k = i & 127;
    sW[r*WSTR + k] = (__bf16)ff_w[i];
  }
  for (int i = tid; i < CC*64; i += 128) {
    int j = i & 63, c = i >> 6;
    sX[j*WSTR + c] = (__bf16)featnew[(size_t)b*CC*n1 + (size_t)c*n1 + n0 + j];
  }
  __syncthreads();

  int half = lane >> 4, sj = lane & 15;
  int col = wv * 16 + sj;
  for (int mt = 0; mt < 8; ++mt) {
    v8f acc = {};
    for (int kt = 0; kt < 4; ++kt) {
      int koff = kt * 32;
      const __bf16* pA = sW + (mt*16 + sj)*WSTR + koff + half*8;
      v16bf a = ld_frag16(pA, pA + 16);
      const __bf16* pB = sX + col*WSTR + koff + half*8;
      v16bf bf = ld_frag16(pB, pB + 16);
      acc = __builtin_amdgcn_wmma_f32_16x16x32_bf16(false, a, false, bf,
                                                    (short)0, acc, false, false);
    }
    #pragma unroll
    for (int r = 0; r < 8; ++r) {
      int M = mt*16 + half*8 + r;
      hbuf[(size_t)b*CC*n1 + (size_t)M*n1 + n0 + col] = acc[r] + ff_b[M];
    }
  }
}

// ---------------------------------------------------------------------------
// Kernel 5: final GroupNorm(8) + leaky on hbuf -> output.
// ---------------------------------------------------------------------------
__global__ __launch_bounds__(256) void gn_final_kernel(
    const float* __restrict__ hbuf, float* __restrict__ out,
    const float* __restrict__ gamma, const float* __restrict__ beta, int N)
{
  int b = blockIdx.x >> 3;
  int g = blockIdx.x & 7;
  int tid = threadIdx.x;
  int total = 16 * N;
  const float* H = hbuf + (size_t)b * CC * N;

  float sum = 0.f, sq = 0.f;
  for (int e = tid; e < total; e += 256) {
    int cl = e / N;
    int n  = e - cl * N;
    int c  = g * 16 + cl;
    float h = H[(size_t)c * N + n];
    sum += h; sq += h * h;
  }
  #pragma unroll
  for (int o = 16; o >= 1; o >>= 1) {
    sum += __shfl_xor(sum, o, 32);
    sq  += __shfl_xor(sq,  o, 32);
  }
  __shared__ float rs[8], rq[8];
  __shared__ float smu, srstd;
  if ((tid & 31) == 0) { rs[tid >> 5] = sum; rq[tid >> 5] = sq; }
  __syncthreads();
  if (tid == 0) {
    float S = 0.f, Q = 0.f;
    for (int i = 0; i < 8; ++i) { S += rs[i]; Q += rq[i]; }
    float mu = S / (float)total;
    float var = Q / (float)total - mu * mu;
    smu = mu; srstd = rsqrtf(var + 1e-5f);
  }
  __syncthreads();
  float mu = smu, rstd = srstd;
  for (int e = tid; e < total; e += 256) {
    int cl = e / N;
    int n  = e - cl * N;
    int c  = g * 16 + cl;
    float h = H[(size_t)c * N + n];
    float v = leakyf((h - mu) * rstd * gamma[c] + beta[c]);
    out[(size_t)b * CC * N + (size_t)c * N + n] = v;
  }
}

// ---------------------------------------------------------------------------
extern "C" void kernel_launch(void* const* d_in, const int* in_sizes, int n_in,
                              void* d_out, int out_size, void* d_ws, size_t ws_size,
                              hipStream_t stream) {
  const float* xyz1  = (const float*)d_in[0];
  const float* xyz2  = (const float*)d_in[1];
  const float* feat1 = (const float*)d_in[2];
  const float* feat2 = (const float*)d_in[3];
  const float* pos_w = (const float*)d_in[4];
  const float* pos_b = (const float*)d_in[5];
  const float* pos_g = (const float*)d_in[6];
  const float* pos_be= (const float*)d_in[7];
  const float* ek_w  = (const float*)d_in[8];
  const float* ek_b  = (const float*)d_in[9];
  const float* eq_w  = (const float*)d_in[10];
  const float* eq_b  = (const float*)d_in[11];
  const float* ev1_w = (const float*)d_in[12];
  const float* ev1_b = (const float*)d_in[13];
  const float* ev2_w = (const float*)d_in[14];
  const float* ev2_b = (const float*)d_in[15];
  const float* qk_w  = (const float*)d_in[16];
  const float* v_w   = (const float*)d_in[17];
  const float* ff_w  = (const float*)d_in[18];
  const float* ff_b  = (const float*)d_in[19];
  const float* ff_g  = (const float*)d_in[20];
  const float* ff_be = (const float*)d_in[21];

  const size_t featElems = (size_t)BB * CC * N1S; // 2,097,152
  float* feat1p  = (float*)d_ws;
  float* feat2p  = feat1p + featElems;
  int*   idxbuf  = (int*)(feat2p + featElems);
  float* featnew = (float*)(idxbuf + (size_t)BB * N1S * NSAMP);
  float* hbuf    = featnew + featElems;

  // 1) positional encoding + GN + add (feat1 and feat2)
  posenc_kernel<<<BB * 8, 256, 0, stream>>>(xyz1, feat1, feat1p,
                                            pos_w, pos_b, pos_g, pos_be, N1S);
  posenc_kernel<<<BB * 8, 256, 0, stream>>>(xyz2, feat2, feat2p,
                                            pos_w, pos_b, pos_g, pos_be, N2S);

  // 2) kNN top-16 (xyz2 staged in LDS: 96 KB)
  knn_kernel<<<BB * (N1S / 256), 256, 3 * N2S * sizeof(float), stream>>>(
      xyz1, xyz2, idxbuf, N1S, N2S);

  // 3) fused gather + encoders + WMMA attention
  {
    int nQB = BB * N1S / 4;
    size_t shmem = (size_t)2 * CC * WSTR * 2          // sQK + sVW (padded bf16)
                 + (size_t)4 * 512 * 4                // encoder weights f32
                 + (size_t)4 * (4 * 16 * WSTR * 2 + 256); // per-wave buffers
    atten_kernel<<<512, 128, shmem, stream>>>(
        xyz1, xyz2, feat1p, feat2p, idxbuf,
        ek_w, ek_b, eq_w, eq_b, ev1_w, ev1_b, ev2_w, ev2_b,
        qk_w, v_w, featnew, N1S, N2S, nQB);
  }

  // 4) feed-forward GEMM via WMMA
  {
    size_t shmem = (size_t)CC * WSTR * 2 + (size_t)64 * WSTR * 2;
    ff_gemm_kernel<<<BB * (N1S / 64), 128, shmem, stream>>>(
        featnew, ff_w, ff_b, hbuf, N1S);
  }

  // 5) final GN + leaky into d_out (after xyz1 block)
  float* outf = (float*)d_out;
  gn_final_kernel<<<BB * 8, 256, 0, stream>>>(hbuf, outf + (size_t)BB * 3 * N1S,
                                              ff_g, ff_be, N1S);

  // 6) copy xyz1 into output tuple slot 0
  hipMemcpyAsync(d_out, d_in[0], (size_t)BB * 3 * N1S * sizeof(float),
                 hipMemcpyDeviceToDevice, stream);
}
